// GraphNeuralNetwork_58102317580569
// MI455X (gfx1250) — compile-verified, matrix operand order checked by reading, not run on hardware
//
#include <hip/hip_runtime.h>

// ---------------------------------------------------------------------------
// GNN forward for MI455X (gfx1250, wave32, WMMA).
// B=128 graphs, N=64 nodes, D=64 feat, M=8192, HL=3, OL=2.
// - Adjacency is block-diagonal -> compact the 128 (64x64) blocks to bf16 once.
// - All GEMMs: v_wmma_f32_16x16x32_bf16 with pre-converted bf16 operands.
// - Message-passing GEMM stages both 64x64 panels in LDS (1 workgroup/graph).
// - Branchless selu via hardware v_exp_f32.
// ---------------------------------------------------------------------------

typedef __bf16 bf16_t;
typedef __attribute__((ext_vector_type(16))) __bf16 v16bf;
typedef __attribute__((ext_vector_type(8)))  __bf16 v8bf;
typedef __attribute__((ext_vector_type(8)))  float  v8f;

#define MTOT   8192
#define DD     64
#define BGR    128
#define NN     64
#define BN_EPS 1e-5f

static __device__ __forceinline__ bf16_t f2bf(float f) {
    // round-to-nearest-even fp32 -> bf16
    unsigned u = __builtin_bit_cast(unsigned, f);
    unsigned r = u + 0x7FFFu + ((u >> 16) & 1u);
    unsigned short h = (unsigned short)(r >> 16);
    return __builtin_bit_cast(bf16_t, h);
}

// Branchless selu: exp argument clamped to <=0 (never overflows), one cndmask.
static __device__ __forceinline__ float selu_f(float z) {
    const float scale = 1.0507009873554805f;
    const float alpha = 1.6732632423543772f;
    float e = alpha * (__expf(fminf(z, 0.f)) - 1.f);
    return scale * (z > 0.f ? z : e);
}

// Build one 16x32 WMMA operand register set for this lane from a bf16 row
// (global or LDS). Per ISA layout: elems 0..7 -> K = kt+half*8+j, 8..15 -> +16.
template <typename PTR>
static __device__ __forceinline__ v16bf load_op(PTR row, int kt, int half) {
    v8bf lo = *(const v8bf*)(row + kt + half * 8);        // 16B aligned
    v8bf hi = *(const v8bf*)(row + kt + 16 + half * 8);   // 16B aligned
    return __builtin_shufflevector(lo, hi, 0, 1, 2, 3, 4, 5, 6, 7,
                                           8, 9, 10, 11, 12, 13, 14, 15);
}

// ---------------------------------------------------------------------------
// x[m][d] = embed[fingerprints[m]][d]  (fp32 master + bf16 mirror)
__global__ void k_embed(const int* __restrict__ fp, const float* __restrict__ embed,
                        float* __restrict__ x, bf16_t* __restrict__ xbf) {
    int idx = blockIdx.x * blockDim.x + threadIdx.x;   // < MTOT*DD
    int m = idx >> 6, d = idx & 63;
    float v = embed[(size_t)fp[m] * DD + d];
    x[idx] = v;
    xbf[idx] = f2bf(v);
}

// Generic fp32 -> bf16 convert (weights)
__global__ void k_f2bf(const float* __restrict__ in, bf16_t* __restrict__ out) {
    int idx = blockIdx.x * blockDim.x + threadIdx.x;
    out[idx] = f2bf(in[idx]);
}

// Compact the 128 diagonal 64x64 adjacency blocks into dense bf16.
__global__ void k_cvt_adj(const float* __restrict__ Adj, bf16_t* __restrict__ adjB) {
    int idx = blockIdx.x * blockDim.x + threadIdx.x;   // < BGR*NN*NN
    int g = idx >> 12;
    int i = (idx >> 6) & 63;
    int j = idx & 63;
    adjB[idx] = f2bf(Adj[(size_t)(g * NN + i) * MTOT + g * NN + j]);
}

// ---------------------------------------------------------------------------
// S = selu(Xb @ Wb^T + bias), Xb:[Mrows,64] bf16, Wb:[64,64] bf16.
// One wave per 16x16 tile; K=64 -> 2 x wmma_f32_16x16x32_bf16.
// mode 0: store S transposed as bf16 into hT[n][m] (row stride MTOT).
// mode 1: store S as f32 into out[m][n] + accumulate BN stat atomics.
__global__ void k_gemm_selu(const bf16_t* __restrict__ Xb, const bf16_t* __restrict__ Wb,
                            const float* __restrict__ bias,
                            bf16_t* __restrict__ hT, float* __restrict__ out,
                            float* __restrict__ sum, float* __restrict__ sumsq,
                            int mode) {
    int gtid = blockIdx.x * blockDim.x + threadIdx.x;
    int wave = gtid >> 5;
    int lane = threadIdx.x & 31;
    int mt = wave >> 2;                    // D/16 = 4 n-tiles
    int nt = wave & 3;
    int r = lane & 15, half = lane >> 4;
    int mbase = mt * 16, nbase = nt * 16;

    const bf16_t* xr = Xb + (size_t)(mbase + r) * DD;   // A: row m = r
    const bf16_t* wr = Wb + (size_t)(nbase + r) * DD;   // B: col n = r (W row)

    v8f c = {};
#pragma unroll
    for (int kt = 0; kt < DD; kt += 32) {
        v16bf a = load_op(xr, kt, half);
        v16bf b = load_op(wr, kt, half);
        c = __builtin_amdgcn_wmma_f32_16x16x32_bf16(false, a, false, b,
                                                    (short)0, c, false, false);
    }

    int n = nbase + r;
    float bn = bias[n];
    if (mode == 0) {
        v8bf pack;
#pragma unroll
        for (int v = 0; v < 8; ++v)
            pack[v] = f2bf(selu_f(c[v] + bn));
        *(v8bf*)(hT + (size_t)n * MTOT + mbase + half * 8) = pack;  // one b128 store
    } else {
        float ls = 0.f, lq = 0.f;
#pragma unroll
        for (int v = 0; v < 8; ++v) {
            int m = mbase + half * 8 + v;
            float s = selu_f(c[v] + bn);
            out[(size_t)m * DD + n] = s;
            ls += s; lq += s * s;
        }
        atomicAdd(&sum[n], ls);
        atomicAdd(&sumsq[n], lq);
    }
}

// ---------------------------------------------------------------------------
// Per-graph t = A_block @ h + x (in-place into x) + BN stat accumulation.
// One workgroup (16 waves = 512 threads) per graph. Both 64x64 bf16 panels are
// staged in LDS once, then each wave computes one 16x16 tile from LDS.
__global__ void __launch_bounds__(512)
k_adj_gemm(const bf16_t* __restrict__ adjB, const bf16_t* __restrict__ hT,
           float* __restrict__ x,
           float* __restrict__ sum, float* __restrict__ sumsq) {
    __shared__ bf16_t ldsA[NN * NN];   // adjacency block, row-major      (8 KB)
    __shared__ bf16_t ldsH[NN * NN];   // ldsH[n][k] = h[g*64+k][n]       (8 KB)

    int g = blockIdx.x;
    int tid = threadIdx.x;             // 0..511
    int rowg = g * NN;

    // Stage adjacency block: 4096 bf16 / 512 threads = 8 bf16 (16B) each.
    {
        const v8bf* src = (const v8bf*)(adjB + (size_t)g * NN * NN);
        ((v8bf*)ldsA)[tid] = src[tid];
    }
    // Stage h panel: thread t copies 8 bf16 of feature-row n = t>>3.
    {
        int n = tid >> 3, ch = tid & 7;
        const v8bf* src = (const v8bf*)(hT + (size_t)n * MTOT + rowg);
        *((v8bf*)(ldsH + n * NN) + ch) = src[ch];
    }
    __syncthreads();

    int wave = tid >> 5;               // 0..15 -> (mt, nt)
    int lane = tid & 31;
    int mt = wave >> 2, nt = wave & 3;
    int r = lane & 15, half = lane >> 4;
    int nfeat = nt * 16 + r;

    const bf16_t* ar = ldsA + (mt * 16 + r) * NN;   // A row (LDS)
    const bf16_t* hr = ldsH + nfeat * NN;           // B col n over k (LDS)

    v8f c = {};
#pragma unroll
    for (int kt = 0; kt < NN; kt += 32) {
        v16bf a = load_op(ar, kt, half);
        v16bf b = load_op(hr, kt, half);
        c = __builtin_amdgcn_wmma_f32_16x16x32_bf16(false, a, false, b,
                                                    (short)0, c, false, false);
    }

    float ls = 0.f, lq = 0.f;
#pragma unroll
    for (int v = 0; v < 8; ++v) {
        int m = rowg + mt * 16 + half * 8 + v;
        size_t off = (size_t)m * DD + nfeat;
        float t = c[v] + x[off];             // residual, elementwise-exclusive
        x[off] = t;
        ls += t; lq += t * t;
    }
    atomicAdd(&sum[nfeat], ls);
    atomicAdd(&sumsq[nfeat], lq);
}

// ---------------------------------------------------------------------------
__global__ void k_zero_stats(float* __restrict__ sum, float* __restrict__ sumsq) {
    int t = threadIdx.x;                     // 128 threads
    if (t < 64) sum[t] = 0.f; else sumsq[t - 64] = 0.f;
}

// t = (t - mu) * rsqrt(var + eps); writes fp32 master + bf16 mirror.
__global__ void k_bn_norm(float* __restrict__ t, bf16_t* __restrict__ tbf,
                          const float* __restrict__ sum, const float* __restrict__ sumsq,
                          float invM) {
    int idx = blockIdx.x * blockDim.x + threadIdx.x;   // < Mrows*64
    int n = idx & 63;
    float mu  = sum[n] * invM;
    float var = sumsq[n] * invM - mu * mu;
    float rs  = rsqrtf(var + BN_EPS);
    float v = (t[idx] - mu) * rs;
    t[idx] = v;
    tbf[idx] = f2bf(v);
}

// mol[b][d] = mean over the graph's 64 nodes (fp32 + bf16 mirror)
__global__ void k_pool(const float* __restrict__ x, float* __restrict__ mol,
                       bf16_t* __restrict__ molbf) {
    int idx = blockIdx.x * blockDim.x + threadIdx.x;   // < BGR*DD
    int b = idx >> 6, d = idx & 63;
    const float* xp = x + (size_t)b * NN * DD + d;
    float s = 0.f;
#pragma unroll
    for (int n2 = 0; n2 < NN; ++n2) s += xp[n2 * DD];
    s *= (1.f / NN);
    mol[idx] = s;
    molbf[idx] = f2bf(s);
}

// out[b] = mol[b] . Wp + bp
__global__ void k_final(const float* __restrict__ mol, const float* __restrict__ Wp,
                        const float* __restrict__ bp, float* __restrict__ out) {
    int b = threadIdx.x;                     // 128 threads
    float s = bp[0];
#pragma unroll
    for (int d = 0; d < DD; ++d) s += mol[b * DD + d] * Wp[d];
    out[b] = s;
}

// ---------------------------------------------------------------------------
extern "C" void kernel_launch(void* const* d_in, const int* in_sizes, int n_in,
                              void* d_out, int out_size, void* d_ws, size_t ws_size,
                              hipStream_t stream) {
    const int*   fp    = (const int*)  d_in[0];
    const float* adj   = (const float*)d_in[1];
    const float* embed = (const float*)d_in[2];
    const float* Wf    = (const float*)d_in[3];
    const float* bf    = (const float*)d_in[4];
    const float* Wo    = (const float*)d_in[5];
    const float* bo    = (const float*)d_in[6];
    const float* Wp    = (const float*)d_in[7];
    const float* bp    = (const float*)d_in[8];
    float* out = (float*)d_out;

    char* ws = (char*)d_ws;
    float*  x     = (float*)ws;  ws += (size_t)MTOT * DD * sizeof(float);     // 2 MB
    bf16_t* xbf   = (bf16_t*)ws; ws += (size_t)MTOT * DD * sizeof(bf16_t);    // 1 MB
    bf16_t* hT    = (bf16_t*)ws; ws += (size_t)DD * MTOT * sizeof(bf16_t);    // 1 MB
    bf16_t* adjB  = (bf16_t*)ws; ws += (size_t)BGR * NN * NN * sizeof(bf16_t);// 1 MB
    bf16_t* Wfbf  = (bf16_t*)ws; ws += (size_t)3 * DD * DD * sizeof(bf16_t);
    bf16_t* Wobf  = (bf16_t*)ws; ws += (size_t)2 * DD * DD * sizeof(bf16_t);
    float*  sum   = (float*)ws;  ws += 64 * sizeof(float);
    float*  sumsq = (float*)ws;  ws += 64 * sizeof(float);
    float*  mol   = (float*)ws;  ws += (size_t)BGR * DD * sizeof(float);
    float*  mol2  = (float*)ws;  ws += (size_t)BGR * DD * sizeof(float);
    bf16_t* molbf = (bf16_t*)ws; ws += (size_t)BGR * DD * sizeof(bf16_t);

    // one-shot operand preparation (bf16 everywhere WMMA reads)
    k_embed<<<(MTOT * DD) / 256, 256, 0, stream>>>(fp, embed, x, xbf);
    k_f2bf<<<(3 * DD * DD) / 256, 256, 0, stream>>>(Wf, Wfbf);
    k_f2bf<<<(2 * DD * DD) / 256, 256, 0, stream>>>(Wo, Wobf);
    k_cvt_adj<<<(BGR * NN * NN) / 256, 256, 0, stream>>>(adj, adjB);

    // 3 message-passing layers
    for (int i = 0; i < 3; ++i) {
        k_gemm_selu<<<(MTOT / 16 * 4 * 32) / 256, 256, 0, stream>>>(
            xbf, Wfbf + (size_t)i * DD * DD, bf + i * DD,
            hT, nullptr, nullptr, nullptr, /*mode=*/0);
        k_zero_stats<<<1, 128, 0, stream>>>(sum, sumsq);
        k_adj_gemm<<<BGR, 512, 0, stream>>>(adjB, hT, x, sum, sumsq);
        k_bn_norm<<<(MTOT * DD) / 256, 256, 0, stream>>>(x, xbf, sum, sumsq, 1.f / MTOT);
    }

    // per-graph mean pool
    k_pool<<<(BGR * DD) / 256, 256, 0, stream>>>(x, mol, molbf);

    // 2 output MLP layers: BN(selu(mol @ Wo[j]^T + bo[j]))
    float* cur = mol; float* nxt = mol2;
    for (int j = 0; j < 2; ++j) {
        k_zero_stats<<<1, 128, 0, stream>>>(sum, sumsq);
        k_gemm_selu<<<(BGR / 16 * 4 * 32) / 256, 256, 0, stream>>>(
            molbf, Wobf + (size_t)j * DD * DD, bo + j * DD,
            nullptr, nxt, sum, sumsq, /*mode=*/1);
        k_bn_norm<<<(BGR * DD) / 256, 256, 0, stream>>>(nxt, molbf, sum, sumsq, 1.f / BGR);
        float* tmp = cur; cur = nxt; nxt = tmp;
    }

    // out = mol @ Wp + bp
    k_final<<<1, 128, 0, stream>>>(cur, Wp, bp, out);
}
